// AttentionBlock_41747082117411
// MI455X (gfx1250) — matrix operand order repaired and checked by reading, not compiled, not run on hardware
//
#include <hip/hip_runtime.h>
#include <hip/hip_bf16.h>
#include <math.h>

// ---------------------------------------------------------------------------
// AttentionBlock for MI455X (gfx1250, wave32, WMMA + TDM).
//   x:[4,512,64,64] f32, GroupNorm(32), QKV 1x1, softmax(QK^T/sqrt(C))V, proj.
// ~172 GFLOP vs <200MB traffic @23.3TB/s -> strongly compute bound: all
// matmuls run through v_wmma_f32_16x16x32_f16 (f32 accumulate).  GEMM tiles
// are staged into double-buffered LDS by the Tensor Data Mover, software-
// pipelined on TENSORcnt (issue tile i+1, wait <=2 so tile i is resident
// while tile i+1 DMAs behind the WMMA work).
// ---------------------------------------------------------------------------

#define CH    512
#define NTOK  4096           // 64*64
#define BATCH 4
#define MROWS (BATCH * NTOK) // 16384, batch folded into GEMM M
#define GROUPS 32
#define CPG   (CH / GROUPS)  // 16
#define GN_EPS 1e-5f

typedef __attribute__((ext_vector_type(16))) _Float16 v16h;
typedef __attribute__((ext_vector_type(8)))  _Float16 h8;
typedef __attribute__((ext_vector_type(8)))  float    v8f;

#if __has_builtin(__builtin_amdgcn_tensor_load_to_lds)
#define HAVE_TDM 1
#else
#define HAVE_TDM 0
#endif

// ---- WMMA fragment helpers (layouts per CDNA5 ISA 7.12.2, wave32) ----------

// A fragment 16x32 f16: lane L holds row L%16; lanes<16 hold K={0..7,16..23},
// lanes>=16 hold K={8..15,24..31}.  Source row-major, leading dim lda (halves).
__device__ __forceinline__ v16h load_frag_a(const _Float16* p, int lda) {
  const int lane = threadIdx.x & 31;
  const int row  = lane & 15;
  const int kb   = (lane & 16) ? 8 : 0;
  const _Float16* r = p + (size_t)row * lda;
  h8 lo = *(const h8*)(r + kb);
  h8 hi = *(const h8*)(r + 16 + kb);
  v16h a;
#pragma unroll
  for (int i = 0; i < 8; ++i) { a[i] = lo[i]; a[8 + i] = hi[i]; }
  return a;
}

// B fragment 32x16 f16: lane L holds column N=L%16; lanes<16 hold
// K={0..7,16..23}, lanes>=16 hold K={8..15,24..31}.
__device__ __forceinline__ v16h load_frag_b(const _Float16* p, int ldb) {
  const int lane = threadIdx.x & 31;
  const int col  = lane & 15;
  const int kb   = (lane & 16) ? 8 : 0;
  v16h b;
#pragma unroll
  for (int i = 0; i < 8; ++i) {
    b[i]     = p[(size_t)(kb + i) * ldb + col];
    b[8 + i] = p[(size_t)(16 + kb + i) * ldb + col];
  }
  return b;
}

// C/D fragment: VGPR r -> M = r (lanes 0-15) or r+8 (lanes 16-31), N = lane&15.
__device__ __forceinline__ int crow(int r) {
  return ((threadIdx.x & 31) < 16) ? r : (r + 8);
}

__device__ __forceinline__ v8f wmma16(v16h a, v16h b, v8f c) {
  return __builtin_amdgcn_wmma_f32_16x16x32_f16(false, a, false, b,
                                                (short)0, c, false, false);
}

// ---- Tensor Data Mover: 2D tile (2-byte elements) global -> LDS ------------
#if HAVE_TDM
typedef unsigned int u32x4 __attribute__((ext_vector_type(4)));
typedef int          i32x4 __attribute__((ext_vector_type(4)));
typedef int          i32x8 __attribute__((ext_vector_type(8)));

// ISA 10.2: generic LDS addresses truncate to the LDS byte offset.
__device__ __forceinline__ unsigned lds_off(const void* p) {
  return (unsigned)(size_t)p;
}

// Load a [tileD1 rows x tileD0 halves] tile (row stride = stride0 halves)
// into LDS at ldsDst, with per-row padding so the LDS leading dim becomes
// tileD0 + 8*(padAmt+1)/... (padAmt code: DWORDs-1; padIvl code: 2^(v+1) DW).
__device__ __forceinline__ void tdm_load_2d(const void* gsrc, unsigned ldsDst,
                                            unsigned tileD0, unsigned tileD1,
                                            unsigned long long stride0,
                                            unsigned padIvl, unsigned padAmt) {
  const unsigned long long ga = (unsigned long long)gsrc;
  u32x4 g0 = {0u, 0u, 0u, 0u};
  g0[0] = 1u;                                     // count=1, user descriptor
  g0[1] = ldsDst;                                 // lds_addr (bytes)
  g0[2] = (unsigned)ga;                           // global_addr[31:0]
  g0[3] = (unsigned)(ga >> 32) | (2u << 30);      // global_addr[56:32], type=2
  i32x8 g1 = {0, 0, 0, 0, 0, 0, 0, 0};
  g1[0] = (1 << 16)                               // data_size = 2 bytes
        | (1 << 20)                               // pad_enable
        | ((int)(padIvl & 7u) << 22)
        | ((int)(padAmt & 127u) << 25);
  g1[1] = (int)(tileD0 << 16);                    // tensor_dim0 = tileD0
  g1[2] = (int)((tileD0 >> 16) | (tileD1 << 16)); // tensor_dim1 = tileD1
  g1[3] = (int)((tileD1 >> 16) | (tileD0 << 16)); // tile_dim0
  g1[4] = (int)(tileD1 & 0xFFFF);                 // tile_dim1 (tile_dim2 = 0)
  g1[5] = (int)(unsigned)(stride0 & 0xFFFFFFFFull); // tensor_dim0_stride lo
  g1[6] = (int)(unsigned)(stride0 >> 32);           // stride hi (dim1_stride=0)
  i32x4 z4 = {0, 0, 0, 0};
#if defined(__clang_major__) && (__clang_major__ >= 23)
  i32x8 z8 = {0, 0, 0, 0, 0, 0, 0, 0};
  __builtin_amdgcn_tensor_load_to_lds(g0, g1, z4, z4, z8, 0);
#else
  __builtin_amdgcn_tensor_load_to_lds(g0, g1, z4, z4, 0);
#endif
}

__device__ __forceinline__ void tdm_wait0() {
#if __has_builtin(__builtin_amdgcn_s_wait_tensorcnt)
  __builtin_amdgcn_s_wait_tensorcnt(0);
#else
  asm volatile("s_wait_tensorcnt 0x0" ::: "memory");
#endif
}
__device__ __forceinline__ void tdm_wait2() {    // <=2 outstanding: previous
#if __has_builtin(__builtin_amdgcn_s_wait_tensorcnt)
  __builtin_amdgcn_s_wait_tensorcnt(2);          // tile's pair has completed
#else
  asm volatile("s_wait_tensorcnt 0x2" ::: "memory");
#endif
}
#endif // HAVE_TDM

// ---------------------------------------------------------------------------
// Kernel 1: GroupNorm statistics. One block per (b,g); the group slab
// x[b, g*16:(g+1)*16, :, :] is 16*4096 contiguous f32.
// ---------------------------------------------------------------------------
__global__ __launch_bounds__(256) void gn_stats(const float* __restrict__ x,
                                                float* __restrict__ stats) {
  const int bg = blockIdx.x;                       // b*32+g, 0..127
  const float* p = x + (size_t)bg * CPG * NTOK;
  const int nelem = CPG * NTOK;                    // 65536
  float s = 0.f, ss = 0.f;
  for (int i = threadIdx.x; i < nelem; i += 256) {
    float v = p[i];
    s += v; ss += v * v;
  }
  __shared__ float rs[256], rss[256];
  rs[threadIdx.x] = s; rss[threadIdx.x] = ss;
  __syncthreads();
  for (int off = 128; off > 0; off >>= 1) {
    if (threadIdx.x < off) {
      rs[threadIdx.x]  += rs[threadIdx.x + off];
      rss[threadIdx.x] += rss[threadIdx.x + off];
    }
    __syncthreads();
  }
  if (threadIdx.x == 0) {
    float mean = rs[0] / (float)nelem;
    float var  = rss[0] / (float)nelem - mean * mean;
    stats[bg * 2 + 0] = mean;
    stats[bg * 2 + 1] = rsqrtf(var + GN_EPS);
  }
}

// ---------------------------------------------------------------------------
// Kernel 2: apply GroupNorm, write token-major f16  hT[b*N + n][c].
// ---------------------------------------------------------------------------
__global__ __launch_bounds__(256) void gn_apply(const float* __restrict__ x,
                                                const float* __restrict__ stats,
                                                const float* __restrict__ w,
                                                const float* __restrict__ bias,
                                                _Float16* __restrict__ hT) {
  const size_t t = (size_t)blockIdx.x * 256 + threadIdx.x; // 64*B*N threads
  const int n  = (int)(t & (NTOK - 1));
  const int b  = (int)((t >> 12) & (BATCH - 1));
  const int c0 = (int)(t >> 14) * 8;               // 8 channels per thread
  h8 out;
#pragma unroll
  for (int i = 0; i < 8; ++i) {
    const int c = c0 + i;
    const int g = c >> 4;
    const float mean = stats[(b * GROUPS + g) * 2 + 0];
    const float rstd = stats[(b * GROUPS + g) * 2 + 1];
    const float v = x[((size_t)b * CH + c) * NTOK + n];
    out[i] = (_Float16)((v - mean) * rstd * w[c] + bias[c]);
  }
  *(h8*)(hT + ((size_t)b * NTOK + n) * CH + c0) = out;
}

// ---------------------------------------------------------------------------
// Kernel 2b: one-shot weight prep.  Wt[c][o] = (f16) W[o][c].
// ---------------------------------------------------------------------------
__global__ __launch_bounds__(256) void prep_w(const float* __restrict__ W,
                                              _Float16* __restrict__ Wt) {
  const int t  = blockIdx.x * 256 + threadIdx.x;   // 512*64 threads
  const int c  = t >> 6;
  const int o0 = (t & 63) * 8;
  h8 out;
#pragma unroll
  for (int i = 0; i < 8; ++i)
    out[i] = (_Float16)W[(size_t)(o0 + i) * CH + c];
  *(h8*)(Wt + (size_t)c * CH + o0) = out;
}

// ---------------------------------------------------------------------------
// Kernel 3/5: GEMM  out[m][o] = sum_c A[m][c] * Wt[c][o] + bias[o]
//   M = 16384, K = 512, N = 512.  128x128 block tile, K-step 32.
//   TDM path: double-buffered LDS, pipelined on TENSORcnt (wave 0 issues).
//   mode A (outH != 0): store f16 token-major (Q/K/V).
//   mode B (outH == 0): f32 transposed store [b][o][n] with residual x added.
// ---------------------------------------------------------------------------
#define LDA 40    // 32 + 8-half pad (TDM: pad 4 DWORDs every 16 DWORDs)
#define LDB 136   // 128 + 8-half pad (TDM: pad 4 DWORDs every 64 DWORDs)

__global__ __launch_bounds__(256) void gemm512(const _Float16* __restrict__ A,
                                               const _Float16* __restrict__ Wt,
                                               const float* __restrict__ bias,
                                               _Float16* __restrict__ outH,
                                               float* __restrict__ outF,
                                               const float* __restrict__ xres) {
#if HAVE_TDM
  __shared__ __align__(16) _Float16 As[2][128 * LDA];
  __shared__ __align__(16) _Float16 Bs[2][32 * LDB];
#else
  __shared__ __align__(16) _Float16 As[1][128 * LDA];
  __shared__ __align__(16) _Float16 Bs[1][32 * LDB];
#endif

  const int bm = blockIdx.x * 128;
  const int bn = blockIdx.y * 128;
  const int tid = threadIdx.x;
  const int wid = tid >> 5, lane = tid & 31;
  const int wm = wid & 3;   // row block of 32
  const int wn = wid >> 2;  // col block of 64

  v8f acc[2][4];
#pragma unroll
  for (int mi = 0; mi < 2; ++mi)
#pragma unroll
    for (int ni = 0; ni < 4; ++ni) acc[mi][ni] = (v8f)0.0f;

#if HAVE_TDM
  if (wid == 0) {                        // EXEC ignored by TDM: 1 wave issues
    tdm_load_2d(A  + (size_t)bm * CH, lds_off(&As[0][0]),
                /*d0=*/32,  /*d1=*/128, /*stride=*/CH, /*ivl=*/3, /*amt=*/3);
    tdm_load_2d(Wt + (size_t)bn,      lds_off(&Bs[0][0]),
                /*d0=*/128, /*d1=*/32,  /*stride=*/CH, /*ivl=*/5, /*amt=*/3);
  }
#endif

  int buf = 0;
  for (int kt = 0; kt < CH; kt += 32) {
#if HAVE_TDM
    if (wid == 0) {
      if (kt + 32 < CH) {                // prefetch next tile into other buf
        tdm_load_2d(A  + (size_t)bm * CH + (kt + 32), lds_off(&As[buf ^ 1][0]),
                    32, 128, CH, 3, 3);
        tdm_load_2d(Wt + (size_t)(kt + 32) * CH + bn, lds_off(&Bs[buf ^ 1][0]),
                    128, 32, CH, 5, 3);
        tdm_wait2();                     // current tile resident, next in flight
      } else {
        tdm_wait0();
      }
    }
#else
    {   // A tile 128x32 halves: each thread moves 16 contiguous halves.
      const int row = tid >> 1, ch = (tid & 1) * 16;
      const _Float16* g = A + (size_t)(bm + row) * CH + kt + ch;
      if (kt + 32 < CH) __builtin_prefetch(g + 32, 0, 1); // global_prefetch_b8
      h8 x0 = *(const h8*)g;
      h8 x1 = *(const h8*)(g + 8);
      *(h8*)&As[0][row * LDA + ch]     = x0;
      *(h8*)&As[0][row * LDA + ch + 8] = x1;
    }
    {   // B tile 32x128 halves from pre-transposed weights.
      const int krow = tid >> 3, ch = (tid & 7) * 16;
      const _Float16* g = Wt + (size_t)(kt + krow) * CH + bn + ch;
      h8 x0 = *(const h8*)g;
      h8 x1 = *(const h8*)(g + 8);
      *(h8*)&Bs[0][krow * LDB + ch]     = x0;
      *(h8*)&Bs[0][krow * LDB + ch + 8] = x1;
    }
#endif
    __syncthreads();

    v16h bf[4];
#pragma unroll
    for (int ni = 0; ni < 4; ++ni)
      bf[ni] = load_frag_b(&Bs[buf][wn * 64 + ni * 16], LDB);
#pragma unroll
    for (int mi = 0; mi < 2; ++mi) {
      v16h af = load_frag_a(&As[buf][(wm * 32 + mi * 16) * LDA], LDA);
#pragma unroll
      for (int ni = 0; ni < 4; ++ni)
        acc[mi][ni] = wmma16(af, bf[ni], acc[mi][ni]);
    }
    __syncthreads();   // buffer free for the TDM issued next iteration
#if HAVE_TDM
    buf ^= 1;
#endif
  }

  // Epilogue
#pragma unroll
  for (int mi = 0; mi < 2; ++mi) {
#pragma unroll
    for (int ni = 0; ni < 4; ++ni) {
#pragma unroll
      for (int r = 0; r < 8; ++r) {
        const int row = bm + wm * 32 + mi * 16 + crow(r);
        const int col = bn + wn * 64 + ni * 16 + (lane & 15);
        const float v = acc[mi][ni][r] + bias[col];
        if (outH) {
          outH[(size_t)row * CH + col] = (_Float16)v;
        } else {
          const int ntok = row & (NTOK - 1);
          const int b    = row >> 12;
          const size_t addr = ((size_t)b * CH + col) * NTOK + ntok;
          outF[addr] = xres[addr] + v;
        }
      }
    }
  }
}

// ---------------------------------------------------------------------------
// Kernel 4: flash attention.  Per block: 64 query rows, loop over 32-key
// tiles.  S = Q K^T * scale by 8 waves (4x2 of 16x16 tiles, 16 WMMA K-steps),
// online softmax in LDS, then each wave holds a 64x64 slice of the O
// accumulator and does P@V (V B-frags straight from global: Q/K/V ~50MB all
// fit in the 192MB L2, so re-reads are L2 hits).
// ---------------------------------------------------------------------------
#define QT 64
#define KTILE 32
#define LDKT 40   // KT_s leading dim (j-extent 32 + pad)
#define LDS_S 33
#define LDP 40

__global__ __launch_bounds__(256) void attn_kernel(const _Float16* __restrict__ Q,
                                                   const _Float16* __restrict__ K,
                                                   const _Float16* __restrict__ V,
                                                   _Float16* __restrict__ O) {
  __shared__ __align__(16) _Float16 KT_s[CH * LDKT];   // [c][j]  40 KB
  __shared__ float    S_s[QT * LDS_S];                 //          8.25 KB
  __shared__ __align__(16) _Float16 P_s[QT * LDP];     //          5 KB
  __shared__ float    mrow[QT], lrow[QT], arow[QT];

  const int b  = blockIdx.y;
  const int q0 = blockIdx.x * QT;
  const _Float16* Qb = Q + (size_t)b * NTOK * CH;
  const _Float16* Kb = K + (size_t)b * NTOK * CH;
  const _Float16* Vb = V + (size_t)b * NTOK * CH;

  const int tid = threadIdx.x;
  const int wid = tid >> 5, lane = tid & 31;
  const int sm = wid & 3, sn = wid >> 2;        // S tile coords (4 x 2)
  const int colb = wid * 64;                    // this wave's O column slice

  const float scale = 0.044194173824159216f;    // 512^-0.5

  v8f acc[4][4];
#pragma unroll
  for (int mi = 0; mi < 4; ++mi)
#pragma unroll
    for (int ni = 0; ni < 4; ++ni) acc[mi][ni] = (v8f)0.0f;

  if (tid < QT) { mrow[tid] = -INFINITY; lrow[tid] = 0.f; }
  __syncthreads();

  for (int k0 = 0; k0 < NTOK; k0 += KTILE) {
    // ---- load K tile transposed into LDS: KT_s[c][j] = K[k0+j][c] ----
#pragma unroll
    for (int it = 0; it < 8; ++it) {
      const int pr = tid + it * 256;            // 2048 (j, c8) pairs
      const int j  = pr & 31;
      const int c8 = pr >> 5;                   // 0..63
      h8 kv = *(const h8*)(Kb + (size_t)(k0 + j) * CH + c8 * 8);
#pragma unroll
      for (int i = 0; i < 8; ++i)
        KT_s[(c8 * 8 + i) * LDKT + j] = kv[i];
    }
    __syncthreads();

    // ---- S tile: each wave one 16x16 tile of the 64x32 S block ----
    v8f sacc = (v8f)0.0f;
    for (int kc = 0; kc < CH; kc += 32) {
      v16h af = load_frag_a(Qb + (size_t)(q0 + sm * 16) * CH + kc, CH);
      v16h bf = load_frag_b(&KT_s[kc * LDKT + sn * 16], LDKT);
      sacc = wmma16(af, bf, sacc);
    }
#pragma unroll
    for (int r = 0; r < 8; ++r)
      S_s[(sm * 16 + crow(r)) * LDS_S + sn * 16 + (lane & 15)] = sacc[r];
    __syncthreads();

    // ---- online softmax: one thread per query row ----
    if (tid < QT) {
      const float mo = mrow[tid];
      float mx = mo;
      for (int j = 0; j < KTILE; ++j)
        mx = fmaxf(mx, S_s[tid * LDS_S + j] * scale);
      const float al = __expf(mo - mx);         // mo = -inf -> 0 on first tile
      float ls = lrow[tid] * al;
      for (int j = 0; j < KTILE; ++j) {
        const float p = __expf(S_s[tid * LDS_S + j] * scale - mx);
        P_s[tid * LDP + j] = (_Float16)p;
        ls += p;
      }
      mrow[tid] = mx; lrow[tid] = ls; arow[tid] = al;
    }
    __syncthreads();

    // ---- rescale accumulator, then O += P @ V ----
#pragma unroll
    for (int mi = 0; mi < 4; ++mi)
#pragma unroll
      for (int ni = 0; ni < 4; ++ni)
#pragma unroll
        for (int r = 0; r < 8; ++r)
          acc[mi][ni][r] *= arow[mi * 16 + crow(r)];

    v16h bf[4];
#pragma unroll
    for (int ni = 0; ni < 4; ++ni)
      bf[ni] = load_frag_b(Vb + (size_t)k0 * CH + colb + ni * 16, CH);
#pragma unroll
    for (int mi = 0; mi < 4; ++mi) {
      v16h af = load_frag_a(&P_s[(mi * 16) * LDP], LDP);
#pragma unroll
      for (int ni = 0; ni < 4; ++ni)
        acc[mi][ni] = wmma16(af, bf[ni], acc[mi][ni]);
    }
    __syncthreads();   // before KT_s / P_s are overwritten next tile
  }

  // ---- finalize: O = acc / l, store f16 token-major ----
#pragma unroll
  for (int mi = 0; mi < 4; ++mi) {
#pragma unroll
    for (int ni = 0; ni < 4; ++ni) {
#pragma unroll
      for (int r = 0; r < 8; ++r) {
        const int row = mi * 16 + crow(r);
        const int col = colb + ni * 16 + (lane & 15);
        const float v = acc[mi][ni][r] / lrow[row];
        O[((size_t)b * NTOK + q0 + row) * CH + col] = (_Float16)v;
      }
    }
  }
}

// ---------------------------------------------------------------------------
// Host-side launcher.  Workspace layout (~69 MB):
//   [0, 4KB)       group-norm stats (128 x {mean, rstd})
//   [4KB, +2MB)    4 transposed f16 weights (512x512 each)
//   then hT (f16 [16384][512], reused as attention output), Q, K, V.
// ---------------------------------------------------------------------------
extern "C" void kernel_launch(void* const* d_in, const int* in_sizes, int n_in,
                              void* d_out, int out_size, void* d_ws, size_t ws_size,
                              hipStream_t stream) {
  const float* x      = (const float*)d_in[0];
  const float* norm_w = (const float*)d_in[1];
  const float* norm_b = (const float*)d_in[2];
  const float* q_w    = (const float*)d_in[3];
  const float* q_b    = (const float*)d_in[4];
  const float* k_w    = (const float*)d_in[5];
  const float* k_b    = (const float*)d_in[6];
  const float* v_w    = (const float*)d_in[7];
  const float* v_b    = (const float*)d_in[8];
  const float* p_w    = (const float*)d_in[9];
  const float* p_b    = (const float*)d_in[10];

  char* ws = (char*)d_ws;
  const size_t WB = (size_t)CH * CH * sizeof(_Float16);    // 512 KB
  const size_t HB = (size_t)MROWS * CH * sizeof(_Float16); // 16 MB
  float*     stats = (float*)ws;
  _Float16*  Wq    = (_Float16*)(ws + 4096);
  _Float16*  Wk    = (_Float16*)(ws + 4096 + 1 * WB);
  _Float16*  Wv    = (_Float16*)(ws + 4096 + 2 * WB);
  _Float16*  Wp    = (_Float16*)(ws + 4096 + 3 * WB);
  char*      big   = ws + 4096 + 4 * WB;
  _Float16*  hT    = (_Float16*)(big);
  _Float16*  Qb    = (_Float16*)(big + 1 * HB);
  _Float16*  Kb    = (_Float16*)(big + 2 * HB);
  _Float16*  Vb    = (_Float16*)(big + 3 * HB);
  _Float16*  attnO = hT;  // hT dead after QKV projections

  gn_stats<<<dim3(BATCH * GROUPS), 256, 0, stream>>>(x, stats);
  gn_apply<<<dim3((64 * MROWS) / 256), 256, 0, stream>>>(x, stats, norm_w, norm_b, hT);

  prep_w<<<dim3(128), 256, 0, stream>>>(q_w, Wq);
  prep_w<<<dim3(128), 256, 0, stream>>>(k_w, Wk);
  prep_w<<<dim3(128), 256, 0, stream>>>(v_w, Wv);
  prep_w<<<dim3(128), 256, 0, stream>>>(p_w, Wp);

  dim3 gg(MROWS / 128, CH / 128);  // 128 x 4
  gemm512<<<gg, 256, 0, stream>>>(hT, Wq, q_b, Qb, nullptr, nullptr);
  gemm512<<<gg, 256, 0, stream>>>(hT, Wk, k_b, Kb, nullptr, nullptr);
  gemm512<<<gg, 256, 0, stream>>>(hT, Wv, v_b, Vb, nullptr, nullptr);

  attn_kernel<<<dim3(NTOK / QT, BATCH), 256, 0, stream>>>(Qb, Kb, Vb, attnO);

  gemm512<<<gg, 256, 0, stream>>>(attnO, Wp, p_b, nullptr, (float*)d_out, x);
}
